// SchNet_44358422233616
// MI455X (gfx1250) — matrix operand order repaired and checked by reading, not compiled, not run on hardware
//
#include <hip/hip_runtime.h>

// ---------------- problem constants ----------------
#define NMOL   2048
#define NATOM  24
#define NROWP  32       // node-side row padding (2 x 16-row WMMA tiles)
#define HDIM   128
#define NPAIR  576      // 24*24 pairs incl. masked diagonal
#define NTILE  36       // 576 / 16
#define NLAYER 4

typedef __attribute__((ext_vector_type(16))) __bf16 bf16x16;
typedef __attribute__((ext_vector_type(8)))  __bf16 bf16x8;
typedef __attribute__((ext_vector_type(8)))  float  f32x8;

// Shifted softplus via hardware transcendentals (v_exp_f32 / v_log_f32)
__device__ __forceinline__ float ssp_f(float x) {
    return fmaxf(x, 0.0f) + __logf(1.0f + __expf(-fabsf(x))) - 0.69314718055994531f;
}

__device__ __forceinline__ f32x8 wmma_bf16(bf16x16 a, bf16x16 b, f32x8 c) {
    return __builtin_amdgcn_wmma_f32_16x16x32_bf16(false, a, false, b, (short)0, c, false, false);
}

// A fragment (16x32 bf16) from row-major bf16 tile, stride HDIM.
// lane row = lane&15; halves 0..7 -> K=kb..kb+7, 8..15 -> K=kb+16..kb+23, kb=(lane&16)?8:0
__device__ __forceinline__ bf16x16 a_frag_bf(const __bf16* tile, int kc, int lane) {
    const __bf16* p = tile + (lane & 15) * HDIM + kc * 32 + ((lane & 16) ? 8 : 0);
    bf16x8 lo = *(const bf16x8*)(p);
    bf16x8 hi = *(const bf16x8*)(p + 16);
    bf16x16 r;
#pragma unroll
    for (int i = 0; i < 8; ++i) { r[i] = lo[i]; r[i + 8] = hi[i]; }
    return r;
}

// A fragment from f32 row-major source (rows padded to NROWP, no guards)
__device__ __forceinline__ bf16x16 a_frag_f32(const float* src, int r0, int kc, int lane) {
    const float* p = src + (r0 + (lane & 15)) * HDIM + kc * 32 + ((lane & 16) ? 8 : 0);
    bf16x16 out;
#pragma unroll
    for (int i = 0; i < 8; ++i) { out[i] = (__bf16)p[i]; out[i + 8] = (__bf16)p[i + 16]; }
    return out;
}

// B fragment (32x16 bf16) from row-major bf16 weight [128][128] in LDS.
__device__ __forceinline__ bf16x16 b_frag_bf(const __bf16* W, int kc, int j, int lane) {
    int k = kc * 32 + ((lane & 16) ? 8 : 0);
    const __bf16* p = W + k * HDIM + j * 16 + (lane & 15);
    bf16x16 r;
#pragma unroll
    for (int i = 0; i < 8; ++i) { r[i] = p[i * HDIM]; r[i + 8] = p[(i + 16) * HDIM]; }
    return r;
}

// B fragment from row-major f32 weight with arbitrary column stride (converted on the fly)
__device__ __forceinline__ bf16x16 b_frag_f32(const float* W, int stride, int kc, int j, int lane) {
    int k = kc * 32 + ((lane & 16) ? 8 : 0);
    const float* p = W + k * stride + j * 16 + (lane & 15);
    bf16x16 r;
#pragma unroll
    for (int i = 0; i < 8; ++i) { r[i] = (__bf16)p[i * stride]; r[i + 8] = (__bf16)p[(i + 16) * stride]; }
    return r;
}

// LDS layout (dynamic):
//  sEdge : NPAIR*HDIM bf16 = 147456 B
//  sW1   : HDIM*HDIM  bf16 =  32768 B (e_w1)
//  sW2   : HDIM*HDIM  bf16 =  32768 B (e_w2)
//  sStage: NROWP*HDIM bf16 =   8192 B (block-shared GEMM re-staging)
//  sXf   : NROWP*HDIM f32  =  16384 B
//  sH    : NROWP*HDIM f32  =  16384 B
//  sAgg  : NROWP*HDIM f32  =  16384 B
//  sRed  : 32 f32          =    128 B
constexpr unsigned SMEM_BYTES =
    NPAIR * HDIM * 2 + 2 * HDIM * HDIM * 2 + NROWP * HDIM * 2 +
    3 * NROWP * HDIM * 4 + 32 * 4; // 270464 bytes < 320 KB/WGP

__global__ __launch_bounds__(256, 1) void schnet_mol(
    const int* __restrict__ atom_charges, const float* __restrict__ coords,
    const float* __restrict__ emb, const float* __restrict__ node_lin_w,
    const float* __restrict__ e_w1, const float* __restrict__ e_b1,
    const float* __restrict__ e_w2, const float* __restrict__ e_b2,
    const float* __restrict__ n_w1, const float* __restrict__ n_b1,
    const float* __restrict__ n_w2, const float* __restrict__ n_b2,
    const float* __restrict__ g_w1, const float* __restrict__ g_b1,
    const float* __restrict__ g_w2, const float* __restrict__ g_b2,
    float* __restrict__ out)
{
    extern __shared__ char smem[];
    __bf16* sEdge  = (__bf16*)smem;
    __bf16* sW1    = sEdge + NPAIR * HDIM;
    __bf16* sW2    = sW1 + HDIM * HDIM;
    __bf16* sStage = sW2 + HDIM * HDIM;
    float*  sXf    = (float*)(sStage + NROWP * HDIM);
    float*  sH     = sXf + NROWP * HDIM;
    float*  sAgg   = sH + NROWP * HDIM;
    float*  sRed   = sAgg + NROWP * HDIM;

    const int tid  = threadIdx.x;
    const int lane = tid & 31;
    const int wave = tid >> 5;
    const int m    = blockIdx.x;
    const int base = m * NATOM;

    // per-wave work split for node-side GEMMs: 2 row tiles x (8 or 4) col tiles
    const int r0w = (wave >> 2) * 16;   // row tile: waves 0-3 -> rows 0..15, waves 4-7 -> 16..31
    const int j0w = (wave & 3) * 2;     // two 16-col tiles per wave (128-wide outputs)

    // ---------- init ----------
    for (int idx = tid; idx < HDIM * HDIM; idx += 256) {
        sW1[idx] = (__bf16)e_w1[idx];
        sW2[idx] = (__bf16)e_w2[idx];
    }
    for (int idx = tid; idx < NROWP * HDIM; idx += 256) {
        int a = idx >> 7, c = idx & 127;
        sH[idx] = (a < NATOM) ? emb[atom_charges[base + a] * HDIM + c] : 0.0f;
    }
    for (int p = tid; p < NPAIR; p += 256) {
        int s = p / 24, d = p % 24;
        float dx = coords[(base + s) * 3 + 0] - coords[(base + d) * 3 + 0];
        float dy = coords[(base + s) * 3 + 1] - coords[(base + d) * 3 + 1];
        float dz = coords[(base + s) * 3 + 2] - coords[(base + d) * 3 + 2];
        sXf[p] = sqrtf(dx * dx + dy * dy + dz * dz); // sXf reused as dist scratch
    }
    __syncthreads();

    // ---------- RBF expansion ----------
    {
        const float step  = 10.0f / 127.0f;
        const float coeff = -0.5f / (step * step);
        for (int idx = tid; idx < NPAIR * HDIM; idx += 256) {
            int p = idx >> 7, g = idx & 127;
            int s = p / 24, d = p % 24;
            float t = sXf[p] - (float)g * step;
            float v = (s != d) ? __expf(coeff * t * t) : 0.0f;
            sEdge[idx] = (__bf16)v;
        }
    }
    __syncthreads();

    // ---------- interaction layers ----------
    for (int layer = 0; layer < NLAYER; ++layer) {
        // ---- x = h @ node_lin_w : all 8 waves, 16 (row,col) units ----
        {
            bf16x16 a0 = a_frag_f32(sH, r0w, 0, lane);
            bf16x16 a1 = a_frag_f32(sH, r0w, 1, lane);
            bf16x16 a2 = a_frag_f32(sH, r0w, 2, lane);
            bf16x16 a3 = a_frag_f32(sH, r0w, 3, lane);
#pragma unroll
            for (int jj = 0; jj < 2; ++jj) {
                int j = j0w + jj;
                f32x8 acc = {};
                acc = wmma_bf16(a0, b_frag_f32(node_lin_w, HDIM, 0, j, lane), acc);
                acc = wmma_bf16(a1, b_frag_f32(node_lin_w, HDIM, 1, j, lane), acc);
                acc = wmma_bf16(a2, b_frag_f32(node_lin_w, HDIM, 2, j, lane), acc);
                acc = wmma_bf16(a3, b_frag_f32(node_lin_w, HDIM, 3, j, lane), acc);
                int col = j * 16 + (lane & 15);
#pragma unroll
                for (int v = 0; v < 8; ++v) {
                    int row = r0w + v + ((lane >> 4) << 3);
                    sXf[row * HDIM + col] = acc[v]; // padded rows: harmless
                }
            }
        }
        for (int idx = tid; idx < NROWP * HDIM; idx += 256) sAgg[idx] = 0.0f;
        __syncthreads();

        // ---- edge MLP + gather-multiply + scatter-sum (all waves, per-wave tiles) ----
        for (int t = wave; t < NTILE; t += 8) {
            const __bf16* At = sEdge + t * 16 * HDIM;
            bf16x16 a0 = a_frag_bf(At, 0, lane);
            bf16x16 a1 = a_frag_bf(At, 1, lane);
            bf16x16 a2 = a_frag_bf(At, 2, lane);
            bf16x16 a3 = a_frag_bf(At, 3, lane);
            // per-wave private staging: reuse a 16-row slice keyed by wave inside sEdge? No:
            // use registers->LDS round trip through a per-wave slice of sStage is not possible
            // (sStage is 32 rows, shared). Stage through sEdge's own tile would corrupt input.
            // Instead: stage into sStage rows is block-phase only; here each wave owns tile t,
            // so stage via its own output tile region after consuming A (A already in registers).
            __bf16* stg = sEdge + t * 16 * HDIM; // safe: A fragments already loaded to registers
#pragma unroll
            for (int j = 0; j < 8; ++j) {
                f32x8 acc = {};
                acc = wmma_bf16(a0, b_frag_bf(sW1, 0, j, lane), acc);
                acc = wmma_bf16(a1, b_frag_bf(sW1, 1, j, lane), acc);
                acc = wmma_bf16(a2, b_frag_bf(sW1, 2, j, lane), acc);
                acc = wmma_bf16(a3, b_frag_bf(sW1, 3, j, lane), acc);
                float bv = e_b1[j * 16 + (lane & 15)];
#pragma unroll
                for (int v = 0; v < 8; ++v) {
                    int row = v + ((lane >> 4) << 3);
                    stg[row * HDIM + j * 16 + (lane & 15)] = (__bf16)ssp_f(acc[v] + bv);
                }
            }
            bf16x16 t0 = a_frag_bf(stg, 0, lane);
            bf16x16 t1 = a_frag_bf(stg, 1, lane);
            bf16x16 t2 = a_frag_bf(stg, 2, lane);
            bf16x16 t3 = a_frag_bf(stg, 3, lane);
#pragma unroll
            for (int j = 0; j < 8; ++j) {
                f32x8 acc = {};
                acc = wmma_bf16(t0, b_frag_bf(sW2, 0, j, lane), acc);
                acc = wmma_bf16(t1, b_frag_bf(sW2, 1, j, lane), acc);
                acc = wmma_bf16(t2, b_frag_bf(sW2, 2, j, lane), acc);
                acc = wmma_bf16(t3, b_frag_bf(sW2, 3, j, lane), acc);
                int col = j * 16 + (lane & 15);
                float bv = e_b2[col];
#pragma unroll
                for (int v = 0; v < 8; ++v) {
                    int row = v + ((lane >> 4) << 3);
                    int p = t * 16 + row;
                    int s = p / 24, d = p - s * 24;
                    float mask = (s != d) ? 1.0f : 0.0f;          // v_cndmask, no branch
                    float val = (acc[v] + bv) * sXf[s * HDIM + col] * mask;
                    sEdge[p * HDIM + col] = (__bf16)val;
                    atomicAdd(&sAgg[d * HDIM + col], val);        // ds_add_f32 (adds 0 on diag)
                }
            }
        }
        __syncthreads();

        // ---- node MLP phase 1: stage = ssp(agg @ n_w1 + b1), all waves ----
        {
            bf16x16 a0 = a_frag_f32(sAgg, r0w, 0, lane);
            bf16x16 a1 = a_frag_f32(sAgg, r0w, 1, lane);
            bf16x16 a2 = a_frag_f32(sAgg, r0w, 2, lane);
            bf16x16 a3 = a_frag_f32(sAgg, r0w, 3, lane);
#pragma unroll
            for (int jj = 0; jj < 2; ++jj) {
                int j = j0w + jj;
                f32x8 acc = {};
                acc = wmma_bf16(a0, b_frag_f32(n_w1, HDIM, 0, j, lane), acc);
                acc = wmma_bf16(a1, b_frag_f32(n_w1, HDIM, 1, j, lane), acc);
                acc = wmma_bf16(a2, b_frag_f32(n_w1, HDIM, 2, j, lane), acc);
                acc = wmma_bf16(a3, b_frag_f32(n_w1, HDIM, 3, j, lane), acc);
                float bv = n_b1[j * 16 + (lane & 15)];
#pragma unroll
                for (int v = 0; v < 8; ++v) {
                    int row = r0w + v + ((lane >> 4) << 3);
                    sStage[row * HDIM + j * 16 + (lane & 15)] = (__bf16)ssp_f(acc[v] + bv);
                }
            }
        }
        __syncthreads();

        // ---- node MLP phase 2: h += stage @ n_w2 + b2, all waves ----
        {
            const __bf16* St = sStage + r0w * HDIM;
            bf16x16 t0 = a_frag_bf(St, 0, lane);
            bf16x16 t1 = a_frag_bf(St, 1, lane);
            bf16x16 t2 = a_frag_bf(St, 2, lane);
            bf16x16 t3 = a_frag_bf(St, 3, lane);
#pragma unroll
            for (int jj = 0; jj < 2; ++jj) {
                int j = j0w + jj;
                f32x8 acc = {};
                acc = wmma_bf16(t0, b_frag_f32(n_w2, HDIM, 0, j, lane), acc);
                acc = wmma_bf16(t1, b_frag_f32(n_w2, HDIM, 1, j, lane), acc);
                acc = wmma_bf16(t2, b_frag_f32(n_w2, HDIM, 2, j, lane), acc);
                acc = wmma_bf16(t3, b_frag_f32(n_w2, HDIM, 3, j, lane), acc);
                int col = j * 16 + (lane & 15);
                float bv = n_b2[col];
#pragma unroll
                for (int v = 0; v < 8; ++v) {
                    int row = r0w + v + ((lane >> 4) << 3);
                    sH[row * HDIM + col] += acc[v] + bv; // padded rows: harmless
                }
            }
        }
        __syncthreads();
    }

    // ---------- readout: sum_a ssp(h @ g_w1 + b1) @ g_w2 + b2 via WMMA ----------
    if (tid == 0) sRed[0] = 0.0f;
    __syncthreads();
    {
        int j = wave & 3; // g_w1 is [128][64]: 4 col tiles; 2 row tiles -> 8 units = 8 waves
        bf16x16 a0 = a_frag_f32(sH, r0w, 0, lane);
        bf16x16 a1 = a_frag_f32(sH, r0w, 1, lane);
        bf16x16 a2 = a_frag_f32(sH, r0w, 2, lane);
        bf16x16 a3 = a_frag_f32(sH, r0w, 3, lane);
        f32x8 acc = {};
        acc = wmma_bf16(a0, b_frag_f32(g_w1, 64, 0, j, lane), acc);
        acc = wmma_bf16(a1, b_frag_f32(g_w1, 64, 1, j, lane), acc);
        acc = wmma_bf16(a2, b_frag_f32(g_w1, 64, 2, j, lane), acc);
        acc = wmma_bf16(a3, b_frag_f32(g_w1, 64, 3, j, lane), acc);
        int col = j * 16 + (lane & 15);
        float b1v = g_b1[col];
        float w2v = g_w2[col];
        float local = 0.0f;
#pragma unroll
        for (int v = 0; v < 8; ++v) {
            int row = r0w + v + ((lane >> 4) << 3);
            float keep = (row < NATOM) ? 1.0f : 0.0f;  // v_cndmask
            local += keep * ssp_f(acc[v] + b1v) * w2v;
        }
        atomicAdd(&sRed[0], local);
    }
    __syncthreads();
    if (tid == 0) out[m] = sRed[0] + (float)NATOM * g_b2[0];
}

extern "C" void kernel_launch(void* const* d_in, const int* in_sizes, int n_in,
                              void* d_out, int out_size, void* d_ws, size_t ws_size,
                              hipStream_t stream) {
    (void)in_sizes; (void)n_in; (void)out_size; (void)d_ws; (void)ws_size;
    const int*   atom_charges = (const int*)d_in[0];
    const float* coords       = (const float*)d_in[1];
    // d_in[2] node_batch_vec, d_in[3] edge_index: not needed (graph is analytic)
    const float* emb          = (const float*)d_in[4];
    const float* node_lin_w   = (const float*)d_in[5];
    const float* e_w1 = (const float*)d_in[6],  *e_b1 = (const float*)d_in[7];
    const float* e_w2 = (const float*)d_in[8],  *e_b2 = (const float*)d_in[9];
    const float* n_w1 = (const float*)d_in[10], *n_b1 = (const float*)d_in[11];
    const float* n_w2 = (const float*)d_in[12], *n_b2 = (const float*)d_in[13];
    const float* g_w1 = (const float*)d_in[14], *g_b1 = (const float*)d_in[15];
    const float* g_w2 = (const float*)d_in[16], *g_b2 = (const float*)d_in[17];
    float* out = (float*)d_out;

    schnet_mol<<<NMOL, 256, SMEM_BYTES, stream>>>(
        atom_charges, coords, emb, node_lin_w,
        e_w1, e_b1, e_w2, e_b2, n_w1, n_b1, n_w2, n_b2,
        g_w1, g_b1, g_w2, g_b2, out);
}
